// gnnbasedModel_49959059587767
// MI455X (gfx1250) — matrix operand order repaired and checked by reading, not compiled, not run on hardware
//
#include <hip/hip_runtime.h>
#include <hip/hip_bf16.h>
#include <math.h>

typedef __attribute__((ext_vector_type(16))) __bf16 v16bf;
typedef __attribute__((ext_vector_type(8)))  float  v8f;

#define HDIM 256

// ---------------- utility ----------------
__global__ void k_zero(float* __restrict__ p, int n) {
  int i = blockIdx.x * blockDim.x + threadIdx.x;
  if (i < n) p[i] = 0.f;
}

// deg[i] = #edges with dst==i (self-loop +1 added in k_dinv)
__global__ void k_deg(const int* __restrict__ dst, int E, float* __restrict__ deg) {
  int e = blockIdx.x * blockDim.x + threadIdx.x;
  if (e < E) atomicAdd(&deg[dst[e]], 1.f);
}

__global__ void k_dinv(float* __restrict__ deg, int n) {
  int i = blockIdx.x * blockDim.x + threadIdx.x;
  if (i < n) deg[i] = rsqrtf(deg[i] + 1.f);   // deg >= 1 incl. self-loop
}

// ---------------- weight pre-pack into WMMA B-operand layout ----------------
// W: [K x 256] f32 row-major  ->  Wp: per (kt, nt) tile, lane L holds 16
// K-consecutive bf16 values of column nt*16+(L&15), K starting at
// kt*32 + (L>>4)*16 (contiguous k = 16h + s order, same as the A loads).
__global__ void k_packW(const float* __restrict__ W, __bf16* __restrict__ Wp, int K) {
  int t = blockIdx.x * blockDim.x + threadIdx.x;          // one per (tile, lane)
  const int numNT = HDIM / 16;
  int total = (K / 32) * numNT * 32;
  if (t >= total) return;
  int lane = t & 31;
  int tile = t >> 5;
  int nt = tile % numNT;
  int kt = tile / numNT;
  int kbase = kt * 32 + ((lane >> 4) << 4);
  int col   = nt * 16 + (lane & 15);
  __bf16* out = Wp + (size_t)t * 16;
#pragma unroll
  for (int j = 0; j < 16; ++j)
    out[j] = (__bf16)W[(size_t)(kbase + j) * HDIM + col];
}

// ---------------- WMMA GEMM:  Y[N x 256] = X[N x K] @ W ----------------
// One wave per 16x64 output strip; A loaded straight from row-major f32
// (contiguous k = 16h + s order, same as packed B), B from the packed buffer.
__global__ void k_gemm_wmma(const float* __restrict__ X, int ldX, int K,
                            const __bf16* __restrict__ Wp,
                            float* __restrict__ Y, int nRows) {
  int lane = threadIdx.x & 31;
  int wave = threadIdx.x >> 5;
  int tileId = blockIdx.x * 8 + wave;
  int numMT = nRows >> 4;
  if (tileId >= numMT * 4) return;                // 4 strips of 64 cols
  int mBase  = (tileId >> 2) << 4;
  int nStrip = tileId & 3;

  int row = mBase + (lane & 15);
  const float* xrow = X + (size_t)row * ldX;
  int kHalf = (lane >> 4) << 4;                   // 0 or 16

  v8f acc0 = {}, acc1 = {}, acc2 = {}, acc3 = {};
  const v16bf* wp = (const v16bf*)Wp;             // one v16bf per (tile,lane)
  int nt0 = nStrip * 4;
  int numKT = K >> 5;

  for (int kt = 0; kt < numKT; ++kt) {
    // ---- A operand: two contiguous 32B row segments, cvt f32->bf16 ----
    int k0 = (kt << 5) + kHalf;
    const float4* p0 = (const float4*)(xrow + k0);
    const float4* p1 = (const float4*)(xrow + k0 + 8);
    float fa[16];
    *(float4*)(fa + 0)  = p0[0];
    *(float4*)(fa + 4)  = p0[1];
    *(float4*)(fa + 8)  = p1[0];
    *(float4*)(fa + 12) = p1[1];
    v16bf a;
#pragma unroll
    for (int j = 0; j < 16; ++j) a[j] = (__bf16)fa[j];

    // ---- B operands: 4 contiguous 32B loads from packed weights ----
    const v16bf* base = wp + ((size_t)kt * 16 + nt0) * 32 + lane;
    v16bf b0 = base[0];
    v16bf b1 = base[32];
    v16bf b2 = base[64];
    v16bf b3 = base[96];

    acc0 = __builtin_amdgcn_wmma_f32_16x16x32_bf16(false, a, false, b0, (short)0, acc0, false, false);
    acc1 = __builtin_amdgcn_wmma_f32_16x16x32_bf16(false, a, false, b1, (short)0, acc1, false, false);
    acc2 = __builtin_amdgcn_wmma_f32_16x16x32_bf16(false, a, false, b2, (short)0, acc2, false, false);
    acc3 = __builtin_amdgcn_wmma_f32_16x16x32_bf16(false, a, false, b3, (short)0, acc3, false, false);
  }

  // ---- C/D layout: VGPR r, lane L -> M = r + (L>>4)*8, N = L&15 ----
  int mOff = (lane >> 4) << 3;
  int col  = nStrip * 64 + (lane & 15);
#pragma unroll
  for (int r = 0; r < 8; ++r) {
    size_t o = (size_t)(mBase + mOff + r) * HDIM + col;
    Y[o]      = acc0[r];
    Y[o + 16] = acc1[r];
    Y[o + 32] = acc2[r];
    Y[o + 48] = acc3[r];
  }
}

// ---------------- edge scatter: Out[d] += Y[s] * dinv[s]*dinv[d] ----------------
// One wave per edge; lane covers cols lane*4..+3 and +128 (coalesced float4).
__global__ void k_agg(const float* __restrict__ Y,
                      const int* __restrict__ src, const int* __restrict__ dst,
                      const float* __restrict__ dinv,
                      float* __restrict__ Out, int E) {
  int lane = threadIdx.x & 31;
  int e = blockIdx.x * 8 + (threadIdx.x >> 5);
  if (e >= E) return;
  if (e + 4096 < E) {                      // global_prefetch_b8 of edge lists
    __builtin_prefetch(&src[e + 4096], 0, 1);
    __builtin_prefetch(&dst[e + 4096], 0, 1);
  }
  int s = src[e], d = dst[e];
  float w = dinv[s] * dinv[d];
  const float4* ys = (const float4*)(Y + (size_t)s * HDIM) + lane;
  float4 a = ys[0], b = ys[32];
  float* od = Out + (size_t)d * HDIM + lane * 4;
  atomicAdd(od + 0,   a.x * w); atomicAdd(od + 1,   a.y * w);
  atomicAdd(od + 2,   a.z * w); atomicAdd(od + 3,   a.w * w);
  atomicAdd(od + 128, b.x * w); atomicAdd(od + 129, b.y * w);
  atomicAdd(od + 130, b.z * w); atomicAdd(od + 131, b.w * w);
}

// ---------------- epilogue: self-loop + bias + ReLU ----------------
__global__ void k_finalize(float* __restrict__ Out, const float* __restrict__ Y,
                           const float* __restrict__ dinv, const float* __restrict__ b,
                           int total) {
  int idx = blockIdx.x * blockDim.x + threadIdx.x;
  if (idx >= total) return;
  int i = idx >> 8, h = idx & 255;
  float di = dinv[i];
  float v = Out[idx] + Y[idx] * di * di + b[h];
  Out[idx] = v > 0.f ? v : 0.f;
}

// ---------------- global add pool into scores[128 x 512] ----------------
__global__ void k_pool(const float* __restrict__ X, const int* __restrict__ batch,
                       float* __restrict__ pooled, int total, int colOff) {
  int idx = blockIdx.x * blockDim.x + threadIdx.x;
  if (idx >= total) return;
  int i = idx >> 8, h = idx & 255;
  atomicAdd(&pooled[(size_t)batch[i] * 512 + colOff + h], X[idx]);
}

// ---------------- tiny MLP head (128 rows) ----------------
__global__ void k_mlp(const float* __restrict__ A, const float* __restrict__ W,
                      const float* __restrict__ bias, float* __restrict__ Out,
                      int M, int K, int Nout, int sig) {
  int idx = blockIdx.x * blockDim.x + threadIdx.x;
  if (idx >= M * Nout) return;
  int m = idx / Nout, n = idx % Nout;
  float acc = bias[n];
  for (int k = 0; k < K; ++k)
    acc = fmaf(A[(size_t)m * K + k], W[(size_t)k * Nout + n], acc);
  Out[idx] = sig ? 1.f / (1.f + expf(-acc)) : (acc > 0.f ? acc : 0.f);
}

// ---------------- orchestration ----------------
extern "C" void kernel_launch(void* const* d_in, const int* in_sizes, int n_in,
                              void* d_out, int out_size, void* d_ws, size_t ws_size,
                              hipStream_t stream) {
  const float* x1  = (const float*)d_in[0];
  const float* x2  = (const float*)d_in[1];
  const int*   ei1 = (const int*)d_in[2];
  const int*   ei2 = (const int*)d_in[3];
  const int*   ba1 = (const int*)d_in[4];
  const int*   ba2 = (const int*)d_in[5];
  const float* W[3]  = {(const float*)d_in[6], (const float*)d_in[8],  (const float*)d_in[10]};
  const float* bb[3] = {(const float*)d_in[7], (const float*)d_in[9],  (const float*)d_in[11]};
  const float* lw0 = (const float*)d_in[12]; const float* lb0 = (const float*)d_in[13];
  const float* lw1 = (const float*)d_in[14]; const float* lb1 = (const float*)d_in[15];
  const float* lw2 = (const float*)d_in[16]; const float* lb2 = (const float*)d_in[17];
  float* out = (float*)d_out;
  (void)n_in; (void)out_size; (void)ws_size;

  const int FIN = 128;
  int N = in_sizes[0] / FIN;
  int E = in_sizes[2] / 2;

  // workspace carve-out (256B aligned)
  char* ws = (char*)d_ws;
  size_t off = 0;
  auto carve = [&](size_t bytes) -> char* {
    char* p = ws + off; off += (bytes + 255) & ~(size_t)255; return p;
  };
  float*  dinv = (float*) carve((size_t)N * 4);
  __bf16* Wp0  = (__bf16*)carve((size_t)(FIN / 32) * 16 * 32 * 16 * 2);
  __bf16* Wp1  = (__bf16*)carve((size_t)(256 / 32) * 16 * 32 * 16 * 2);
  __bf16* Wp2  = (__bf16*)carve((size_t)(256 / 32) * 16 * 32 * 16 * 2);
  float*  Ybuf = (float*) carve((size_t)N * HDIM * 4);
  float*  Abuf = (float*) carve((size_t)N * HDIM * 4);
  float*  Bbuf = (float*) carve((size_t)N * HDIM * 4);
  float*  pooled = (float*)carve((size_t)128 * 512 * 4);   // [g | cols 0-255 = graph1, 256-511 = graph2]
  float*  h1   = (float*) carve((size_t)128 * 256 * 4);
  float*  h2   = (float*) carve((size_t)128 * 128 * 4);

  // pack weights once (tiny)
  { int tot = (FIN / 32) * 16 * 32; k_packW<<<(tot + 255) / 256, 256, 0, stream>>>(W[0], Wp0, FIN); }
  { int tot = (256 / 32) * 16 * 32; k_packW<<<(tot + 255) / 256, 256, 0, stream>>>(W[1], Wp1, 256); }
  { int tot = (256 / 32) * 16 * 32; k_packW<<<(tot + 255) / 256, 256, 0, stream>>>(W[2], Wp2, 256); }

  k_zero<<<(128 * 512 + 255) / 256, 256, 0, stream>>>(pooled, 128 * 512);

  const __bf16* Wp[3] = {Wp0, Wp1, Wp2};
  int NH = N * HDIM;

  for (int g = 0; g < 2; ++g) {
    const float* x    = g == 0 ? x1 : x2;
    const int*   src  = g == 0 ? ei1 : ei2;
    const int*   dst  = src + E;
    const int*   batch = g == 0 ? ba1 : ba2;

    // symmetric-norm factors
    k_zero<<<(N + 255) / 256, 256, 0, stream>>>(dinv, N);
    k_deg <<<(E + 255) / 256, 256, 0, stream>>>(dst, E, dinv);
    k_dinv<<<(N + 255) / 256, 256, 0, stream>>>(dinv, N);

    const float* act = x;
    int ldX = FIN, K = FIN;
    float* ping = Abuf; float* pong = Bbuf;
    for (int layer = 0; layer < 3; ++layer) {
      int waveTiles = (N / 16) * 4;
      k_gemm_wmma<<<(waveTiles + 7) / 8, 256, 0, stream>>>(act, ldX, K, Wp[layer], Ybuf, N);
      k_zero    <<<(NH + 255) / 256, 256, 0, stream>>>(ping, NH);
      k_agg     <<<(E + 7) / 8, 256, 0, stream>>>(Ybuf, src, dst, dinv, ping, E);
      k_finalize<<<(NH + 255) / 256, 256, 0, stream>>>(ping, Ybuf, dinv, bb[layer], NH);
      act = ping; float* t = ping; ping = pong; pong = t;
      ldX = 256; K = 256;
    }
    k_pool<<<(NH + 255) / 256, 256, 0, stream>>>(act, batch, pooled, NH, g * 256);
  }

  // MLP head: [128,512]->relu 256 ->relu 128 ->sigmoid 1
  k_mlp<<<(128 * 256 + 255) / 256, 256, 0, stream>>>(pooled, lw0, lb0, h1, 128, 512, 256, 0);
  k_mlp<<<(128 * 128 + 255) / 256, 256, 0, stream>>>(h1,     lw1, lb1, h2, 128, 256, 128, 0);
  k_mlp<<<(128 * 1   + 255) / 256, 256, 0, stream>>>(h2,     lw2, lb2, out, 128, 128, 1, 1);
}
